// SelfAttention_76630806495824
// MI455X (gfx1250) — matrix operand order repaired and checked by reading, not compiled, not run on hardware
//
#include <hip/hip_runtime.h>
#include <hip/hip_bf16.h>

typedef __attribute__((ext_vector_type(16))) _Float16     v16h;
typedef __attribute__((ext_vector_type(8)))  float        v8f;
typedef __attribute__((ext_vector_type(4)))  unsigned int v4u;
typedef __attribute__((ext_vector_type(8)))  int          v8i;
typedef __attribute__((ext_vector_type(4)))  int          v4i;

#define LN_EPS 1e-5f
constexpr int B_ = 8, L_ = 1000, D_ = 512, H_ = 8, DK = 64;
constexpr int LP = 1024;          // L padded to multiple of 32
constexpr int BL = B_ * L_;       // 8000
constexpr int BH = B_ * H_;       // 64
constexpr int NREL = 2 * L_ - 1;  // 1999

__device__ inline v8f wmma_f16(v16h a, v16h b, v8f c) {
  // (neg_a, A, neg_b, B, c_mod, C, reuse_a, reuse_b)
  return __builtin_amdgcn_wmma_f32_16x16x32_f16(false, a, false, b, (short)0, c,
                                                false, false);
}

__device__ inline v8f v8f_zero() {
  v8f z = {0.f, 0.f, 0.f, 0.f, 0.f, 0.f, 0.f, 0.f};
  return z;
}

// A fragment 16x32 (MxK) f16, row-major source.
// lane l -> m = l&15, kh = l>>4; elem e<8 -> K = k0+kh*8+e,
// elem e>=8 -> K = k0+16+kh*8+(e-8).
__device__ inline v16h load_a_f16(const _Float16* __restrict__ src, int ld,
                                  int row0, int k0, int ln16, int kh) {
  const _Float16* p = src + (size_t)(row0 + ln16) * ld + k0 + kh * 8;
  v16h a;
#pragma unroll
  for (int e = 0; e < 8; ++e) { a[e] = p[e]; a[e + 8] = p[e + 16]; }
  return a;
}

// B fragment 32x16 (KxN) f16: lane l -> n = l&15, kh = l>>4;
// elem e<8 -> K = k0 + kh*8 + e, elem e>=8 -> K = k0 + 16 + kh*8 + (e-8). colbase = K-contiguous data for col n.
__device__ inline v16h load_b_f16(const _Float16* __restrict__ colbase, int k0,
                                  int kh) {
  const _Float16* p = colbase + k0 + kh * 8;
  v16h b;
#pragma unroll
  for (int e = 0; e < 8; ++e) { b[e] = p[e]; b[e + 8] = p[e + 16]; }
  return b;
}

// ---------------------------------------------------------------------------
// TDM: DMA one (32 x 64) f16 tile from global memory into LDS.
// D# group0: count=1, lds_addr, global_addr(57b), type=2 ("image").
// D# group1: data_size=2B, tensor_dim0=64, tensor_dim1=1024, tile=64x32,
//            dim0_stride=64.  Groups 2/3 zero (2-D, no iterate/gather).
// ---------------------------------------------------------------------------
__device__ inline void tdm_load_tile_32x64(const _Float16* gsrc,
                                           _Float16* lds_dst) {
  unsigned long long ga = (unsigned long long)(uintptr_t)gsrc;
  unsigned lds_off = (unsigned)(uintptr_t)lds_dst;  // flat low 32 == LDS offset
  v4u g0;
  g0[0] = 1u;                                        // count=1, user mode
  g0[1] = lds_off;                                   // lds_addr (bytes)
  g0[2] = (unsigned)(ga & 0xffffffffu);              // global_addr[31:0]
  g0[3] = (unsigned)((ga >> 32) & 0x1ffffffu)        // global_addr[56:32]
          | (2u << 30);                              // type=2
  v8i g1;
  g1[0] = 0x10000;            // [17:16] data_size = 1 -> 2 bytes
  g1[1] = (64 << 16);         // tensor_dim0[15:0] in [31:16] = 64
  g1[2] = (1024 << 16);       // tensor_dim1[15:0] in [31:16] = 1024
  g1[3] = (64 << 16);         // tile_dim0 in [31:16] = 64
  g1[4] = 32;                 // tile_dim1 = 32 rows (tile_dim2 = 0)
  g1[5] = 64;                 // tensor_dim0_stride[31:0] = 64 elements
  g1[6] = 0;                  // stride hi / dim1_stride lo
  g1[7] = 0;                  // dim1_stride hi
  v4i z4 = {0, 0, 0, 0};
  v8i z8 = {0, 0, 0, 0, 0, 0, 0, 0};
  __builtin_amdgcn_tensor_load_to_lds(g0, g1, z4, z4, z8, 0);
}

// ---------------------------------------------------------------------------
// 0) one-shot f32 -> f16 conversion (weights, rel_pos_embed)
// ---------------------------------------------------------------------------
__global__ __launch_bounds__(256) void cvt_kernel(const float* __restrict__ src,
                                                  _Float16* __restrict__ dst,
                                                  int n) {
  int i = blockIdx.x * 256 + threadIdx.x;
  if (i < n) dst[i] = (_Float16)src[i];
}

// ---------------------------------------------------------------------------
// 1) LayerNorm -> xn (f16).  One wave per row of 512.
// ---------------------------------------------------------------------------
__global__ __launch_bounds__(256) void ln_kernel(
    const float* __restrict__ x, const float* __restrict__ gamma,
    const float* __restrict__ beta, _Float16* __restrict__ xn) {
  const int lane = threadIdx.x & 31, wave = threadIdx.x >> 5;
  const int row = blockIdx.x * 8 + wave;
  const float* xr = x + (size_t)row * D_;
  float v[16], s = 0.f, s2 = 0.f;
#pragma unroll
  for (int e = 0; e < 16; ++e) {
    float t = xr[lane + 32 * e];
    v[e] = t; s += t; s2 += t * t;
  }
#pragma unroll
  for (int m = 1; m < 32; m <<= 1) {
    s  += __shfl_xor(s,  m, 32);
    s2 += __shfl_xor(s2, m, 32);
  }
  const float mu   = s * (1.f / D_);
  const float var  = s2 * (1.f / D_) - mu * mu;
  const float rstd = rsqrtf(var + LN_EPS);
  _Float16* xo = xn + (size_t)row * D_;
#pragma unroll
  for (int e = 0; e < 16; ++e) {
    int idx = lane + 32 * e;
    xo[idx] = (_Float16)((v[e] - mu) * rstd * gamma[idx] + beta[idx]);
  }
}

// ---------------------------------------------------------------------------
// 2) Zero the 24 padded rows of Q/K/V.
// ---------------------------------------------------------------------------
__global__ __launch_bounds__(256) void pad_kernel(_Float16* __restrict__ Qo,
                                                  _Float16* __restrict__ Ko,
                                                  _Float16* __restrict__ Vo) {
  int t = blockIdx.x * blockDim.x + threadIdx.x;    // 0 .. 3*64*24*64-1
  const int per = BH * (LP - L_) * DK;              // 98304
  int which = t / per;
  int rem = t % per;
  int bh = rem / ((LP - L_) * DK);
  int rr = rem % ((LP - L_) * DK);
  int pr = rr / DK, d = rr % DK;
  _Float16* dst = which == 0 ? Qo : (which == 1 ? Ko : Vo);
  dst[((size_t)bh * LP + L_ + pr) * DK + d] = (_Float16)0.f;
}

// ---------------------------------------------------------------------------
// 3) QKV projection: out = xn @ W^T + b -> (b,h,Lpad,dk) f16.
//    1 wave per 64x16 tile (4 accums, 4x weight reuse).
//    grid = (125, 4, 3), block = 256.
// ---------------------------------------------------------------------------
__global__ __launch_bounds__(256) void qkv_kernel(
    const _Float16* __restrict__ xn,
    const _Float16* __restrict__ Wq, const _Float16* __restrict__ Wk,
    const _Float16* __restrict__ Wv, const float* __restrict__ bq,
    const float* __restrict__ bk, const float* __restrict__ bv,
    _Float16* __restrict__ Qo, _Float16* __restrict__ Ko,
    _Float16* __restrict__ Vo) {
  const int lane = threadIdx.x & 31, wave = threadIdx.x >> 5;
  const int ln16 = lane & 15, kh = lane >> 4;
  const int r0 = blockIdx.x * 64;
  const int c0 = (blockIdx.y * 8 + wave) * 16;
  const int z = blockIdx.z;
  const _Float16* W    = z == 0 ? Wq : (z == 1 ? Wk : Wv);
  const float*    bias = z == 0 ? bq : (z == 1 ? bk : bv);
  _Float16*       dst  = z == 0 ? Qo : (z == 1 ? Ko : Vo);

  v8f acc[4];
#pragma unroll
  for (int rt = 0; rt < 4; ++rt) acc[rt] = v8f_zero();
  const _Float16* wcol = W + (size_t)(c0 + ln16) * D_;  // B[d][c] = W[c][d]
#pragma unroll 2
  for (int k0 = 0; k0 < D_; k0 += 32) {
    v16h b = load_b_f16(wcol, k0, kh);
#pragma unroll
    for (int rt = 0; rt < 4; ++rt) {
      v16h a = load_a_f16(xn, D_, r0 + rt * 16, k0, ln16, kh);
      acc[rt] = wmma_f16(a, b, acc[rt]);
    }
  }
  const int c = c0 + ln16;
  const int hh = c >> 6, d = c & 63;
  const float bi = bias[c];
#pragma unroll
  for (int rt = 0; rt < 4; ++rt) {
#pragma unroll
    for (int r = 0; r < 8; ++r) {
      int row = r0 + rt * 16 + kh * 8 + r;   // C layout: m = (lane>>4)*8 + r
      int bidx = row / L_;
      int li = row - bidx * L_;
      dst[((size_t)(bidx * H_ + hh) * LP + li) * DK + d] =
          (_Float16)(acc[rt][r] + bi);
    }
  }
}

// ---------------------------------------------------------------------------
// 4) Fused flash attention with relative position bias.
//    grid = (B*H, 8), block = 256; each wave owns a 16-row query tile and
//    streams 32-column K/V tiles (staged by the Tensor Data Mover).
//    14 WMMAs per tile, O-update at full K=32.
// ---------------------------------------------------------------------------
__global__ __launch_bounds__(256) void attn_kernel(
    const _Float16* __restrict__ Q, const _Float16* __restrict__ K,
    const _Float16* __restrict__ V, const _Float16* __restrict__ relh,
    _Float16* __restrict__ ctx) {
  __shared__ _Float16 sK[32 * DK];        // K tile (32 x 64)
  __shared__ _Float16 sV[32 * DK];        // V tile (32 x 64)
  __shared__ float sP[8][16 * 48];        // per-wave P = Q @ relWin^T
  __shared__ float sE[8][16 * 32];        // per-wave exp(S)

  const int lane = threadIdx.x & 31, wave = threadIdx.x >> 5;
  const int ln16 = lane & 15, kh = lane >> 4;
  const int bh = blockIdx.x;
  const int b = bh >> 3, h = bh & 7;
  const int i0 = (blockIdx.y * 8 + wave) * 16;       // 0..1008, exact cover

  const _Float16* Qb = Q + (size_t)bh * LP * DK;
  const _Float16* Kb = K + (size_t)bh * LP * DK;
  const _Float16* Vb = V + (size_t)bh * LP * DK;

  v16h aq0 = load_a_f16(Qb, DK, i0, 0,  ln16, kh);
  v16h aq1 = load_a_f16(Qb, DK, i0, 32, ln16, kh);

  float m_r[8], l_r[8];
  v8f o[4];
#pragma unroll
  for (int r = 0; r < 8; ++r) { m_r[r] = -3e38f; l_r[r] = 0.f; }
#pragma unroll
  for (int t = 0; t < 4; ++t) o[t] = v8f_zero();

  const int mbase = kh * 8;

  for (int j0 = 0; j0 < LP; j0 += 32) {
    __syncthreads();  // previous iteration done reading sK/sV
    if (wave == 0) {  // TDM: async DMA both tiles, wait TENSORcnt==0
      tdm_load_tile_32x64(Kb + (size_t)j0 * DK, sK);
      tdm_load_tile_32x64(Vb + (size_t)j0 * DK, sV);
      __builtin_amdgcn_s_wait_tensorcnt(0);
    }
    if (wave == 1 && j0 + 32 < LP) {     // prefetch next tiles (128B/lane)
      __builtin_prefetch(Kb + (size_t)(j0 + 32) * DK + lane * 64, 0, 1);
      __builtin_prefetch(Vb + (size_t)(j0 + 32) * DK + lane * 64, 0, 1);
    }
    __syncthreads();

    // S = Q @ K^T  (16 x 32, two column tiles)
    v8f s0 = v8f_zero(), s1 = v8f_zero();
    {
      const _Float16* kc0 = sK + ln16 * DK;
      const _Float16* kc1 = sK + (16 + ln16) * DK;
      s0 = wmma_f16(aq0, load_b_f16(kc0, 0,  kh), s0);
      s0 = wmma_f16(aq1, load_b_f16(kc0, 32, kh), s0);
      s1 = wmma_f16(aq0, load_b_f16(kc1, 0,  kh), s1);
      s1 = wmma_f16(aq1, load_b_f16(kc1, 32, kh), s1);
    }

    // P = Q @ relWindow^T : window t0..t0+47 covers all (j-i+L-1) of tile
    const int t0 = j0 - i0 + (L_ - 1) - 15;
    v8f P[3];
#pragma unroll
    for (int pt = 0; pt < 3; ++pt) {
      int t = t0 + pt * 16 + ln16;
      t = t < 0 ? 0 : (t > NREL - 1 ? NREL - 1 : t);
      const _Float16* rb = relh + (size_t)t * DK;
      P[pt] = v8f_zero();
      P[pt] = wmma_f16(aq0, load_b_f16(rb, 0,  kh), P[pt]);
      P[pt] = wmma_f16(aq1, load_b_f16(rb, 32, kh), P[pt]);
    }

    // spill P (16x48 per wave), then gather skewed diagonal pos[m][n-m+15]
#pragma unroll
    for (int r = 0; r < 8; ++r) {
      float* prow = &sP[wave][(mbase + r) * 48];
      prow[ln16]      = P[0][r];
      prow[16 + ln16] = P[1][r];
      prow[32 + ln16] = P[2][r];
    }
    __syncthreads();

    // online softmax update over 32 columns
#pragma unroll
    for (int r = 0; r < 8; ++r) {
      const int m = mbase + r;
      const float* prow = &sP[wave][m * 48];
      float sv0 = s0[r] * 0.125f + prow[ln16 - m + 15];
      float sv1 = s1[r] * 0.125f + prow[ln16 - m + 31];
      if (j0 + ln16 >= L_)      sv0 = -1e30f;
      if (j0 + 16 + ln16 >= L_) sv1 = -1e30f;
      float mx = fmaxf(sv0, sv1);
      mx = fmaxf(mx, __shfl_xor(mx, 1, 32));
      mx = fmaxf(mx, __shfl_xor(mx, 2, 32));
      mx = fmaxf(mx, __shfl_xor(mx, 4, 32));
      mx = fmaxf(mx, __shfl_xor(mx, 8, 32));
      float mn = fmaxf(m_r[r], mx);
      float sc = expf(m_r[r] - mn);
      float p0 = expf(sv0 - mn);
      float p1 = expf(sv1 - mn);
      float rs = p0 + p1;
      rs += __shfl_xor(rs, 1, 32);
      rs += __shfl_xor(rs, 2, 32);
      rs += __shfl_xor(rs, 4, 32);
      rs += __shfl_xor(rs, 8, 32);
      l_r[r] = l_r[r] * sc + rs;
      m_r[r] = mn;
      o[0][r] *= sc; o[1][r] *= sc; o[2][r] *= sc; o[3][r] *= sc;
      sE[wave][m * 32 + ln16]      = p0;
      sE[wave][m * 32 + 16 + ln16] = p1;
    }
    __syncthreads();

    // O += exp(S) @ V : A = 16x32 (full K), B = V tile (32x64)
    v16h ap;
    {
      const float* erow = &sE[wave][ln16 * 32];
#pragma unroll
      for (int e = 0; e < 8; ++e) {
        ap[e]     = (_Float16)erow[kh * 8 + e];        // K = kh*8+e
        ap[e + 8] = (_Float16)erow[16 + kh * 8 + e];   // K = 16+kh*8+e
      }
    }
#pragma unroll
    for (int t = 0; t < 4; ++t) {
      const int dcol = t * 16 + ln16;
      v16h bv;
#pragma unroll
      for (int e = 0; e < 16; ++e)                     // K = kh*8+e (e<8), 16+kh*8+(e-8) (e>=8)
        bv[e] = sV[((e < 8 ? 0 : 8) + kh * 8 + e) * DK + dcol];
      o[t] = wmma_f16(ap, bv, o[t]);
    }
  }

  // finalize and store context (f16, (B,L,D) with D = h*64+d)
#pragma unroll
  for (int r = 0; r < 8; ++r) {
    const int li = i0 + mbase + r;
    if (li < L_) {
      const float inv = 1.f / l_r[r];
      _Float16* crow = ctx + ((size_t)(b * L_ + li)) * D_ + h * DK;
#pragma unroll
      for (int t = 0; t < 4; ++t)
        crow[t * 16 + ln16] = (_Float16)(o[t][r] * inv);
    }
  }
}

// ---------------------------------------------------------------------------
// 5) Output projection + bias + residual -> f32 out. 64x16 tile per wave.
// ---------------------------------------------------------------------------
__global__ __launch_bounds__(256) void out_kernel(
    const _Float16* __restrict__ ctx, const _Float16* __restrict__ Wo,
    const float* __restrict__ bo, const float* __restrict__ x,
    float* __restrict__ out) {
  const int lane = threadIdx.x & 31, wave = threadIdx.x >> 5;
  const int ln16 = lane & 15, kh = lane >> 4;
  const int r0 = blockIdx.x * 64;
  const int c0 = (blockIdx.y * 8 + wave) * 16;

  v8f acc[4];
#pragma unroll
  for (int rt = 0; rt < 4; ++rt) acc[rt] = v8f_zero();
  const _Float16* wcol = Wo + (size_t)(c0 + ln16) * D_;
#pragma unroll 2
  for (int k0 = 0; k0 < D_; k0 += 32) {
    v16h b = load_b_f16(wcol, k0, kh);
#pragma unroll
    for (int rt = 0; rt < 4; ++rt) {
      v16h a = load_a_f16(ctx, D_, r0 + rt * 16, k0, ln16, kh);
      acc[rt] = wmma_f16(a, b, acc[rt]);
    }
  }
  const int c = c0 + ln16;
  const float bi = bo[c];
#pragma unroll
  for (int rt = 0; rt < 4; ++rt) {
#pragma unroll
    for (int r = 0; r < 8; ++r) {
      const int row = r0 + rt * 16 + kh * 8 + r;
      out[(size_t)row * D_ + c] = acc[rt][r] + bi + x[(size_t)row * D_ + c];
    }
  }
}

// ---------------------------------------------------------------------------
extern "C" void kernel_launch(void* const* d_in, const int* in_sizes, int n_in,
                              void* d_out, int out_size, void* d_ws,
                              size_t ws_size, hipStream_t stream) {
  const float* x     = (const float*)d_in[0];
  const float* Wq    = (const float*)d_in[1];
  const float* bq    = (const float*)d_in[2];
  const float* Wk    = (const float*)d_in[3];
  const float* bk    = (const float*)d_in[4];
  const float* Wv    = (const float*)d_in[5];
  const float* bv    = (const float*)d_in[6];
  const float* Wo    = (const float*)d_in[7];
  const float* bo    = (const float*)d_in[8];
  const float* rel   = (const float*)d_in[9];
  const float* gamma = (const float*)d_in[10];
  const float* beta  = (const float*)d_in[11];
  float* out = (float*)d_out;

  _Float16* w    = (_Float16*)d_ws;
  _Float16* xn   = w;                                   // 8000*512
  _Float16* Qb   = xn + (size_t)BL * D_;                // 64*1024*64 each
  _Float16* Kb   = Qb + (size_t)BH * LP * DK;
  _Float16* Vb   = Kb + (size_t)BH * LP * DK;
  _Float16* ctxb = Vb + (size_t)BH * LP * DK;           // 8000*512
  _Float16* Wqh  = ctxb + (size_t)BL * D_;              // 512*512 each
  _Float16* Wkh  = Wqh + (size_t)D_ * D_;
  _Float16* Wvh  = Wkh + (size_t)D_ * D_;
  _Float16* Woh  = Wvh + (size_t)D_ * D_;
  _Float16* relh = Woh + (size_t)D_ * D_;               // 1999*64
  (void)ws_size; (void)in_sizes; (void)n_in; (void)out_size;

  const int nW = D_ * D_, nR = NREL * DK;
  cvt_kernel<<<(nW + 255) / 256, 256, 0, stream>>>(Wq, Wqh, nW);
  cvt_kernel<<<(nW + 255) / 256, 256, 0, stream>>>(Wk, Wkh, nW);
  cvt_kernel<<<(nW + 255) / 256, 256, 0, stream>>>(Wv, Wvh, nW);
  cvt_kernel<<<(nW + 255) / 256, 256, 0, stream>>>(Wo, Woh, nW);
  cvt_kernel<<<(nR + 255) / 256, 256, 0, stream>>>(rel, relh, nR);

  ln_kernel<<<BL / 8, 256, 0, stream>>>(x, gamma, beta, xn);
  pad_kernel<<<(3 * BH * (LP - L_) * DK) / 256, 256, 0, stream>>>(Qb, Kb, Vb);
  qkv_kernel<<<dim3(BL / 64, D_ / 128, 3), 256, 0, stream>>>(
      xn, Wqh, Wkh, Wvh, bq, bk, bv, Qb, Kb, Vb);
  attn_kernel<<<dim3(BH, 8), 256, 0, stream>>>(Qb, Kb, Vb, relh, ctxb);
  out_kernel<<<dim3(BL / 64, D_ / 128), 256, 0, stream>>>(ctxb, Woh, bo, x, out);
}
